// _GCNDyGraphConv2d_42554535969011
// MI455X (gfx1250) — matrix operand order repaired and checked by reading, not compiled, not run on hardware
//
#include <hip/hip_runtime.h>
#include <hip/hip_bf16.h>
#include <math.h>

// Problem constants (from reference)
#define Bc   8
#define Cc   192      // channels
#define Nc   3136     // H*W = 56*56
#define OUTc 192
#define Kc   9
#define Ec   384      // 2*C
#define Rc   (Bc*Nc)  // 25088 total rows

typedef __attribute__((ext_vector_type(16))) _Float16 v16h;
typedef __attribute__((ext_vector_type(16))) __bf16   v16bf;
typedef __attribute__((ext_vector_type(8)))  float    v8f;
typedef __attribute__((ext_vector_type(4)))  unsigned int u32x4;
typedef __attribute__((ext_vector_type(8)))  int      i32x8;
typedef __attribute__((ext_vector_type(4)))  int      i32x4;

#if defined(__has_builtin)
#if __has_builtin(__builtin_amdgcn_tensor_load_to_lds) && __has_builtin(__builtin_amdgcn_s_wait_tensorcnt)
#define USE_TDM 1
#endif
#endif
#ifndef USE_TDM
#define USE_TDM 0
#endif

// ---------- helpers ----------
__device__ __forceinline__ unsigned short f2bf(float f) {
  unsigned u = __float_as_uint(f);
  u += 0x7FFFu + ((u >> 16) & 1u);           // round-to-nearest-even
  return (unsigned short)(u >> 16);
}

// A fragment (16x32 16-bit, MxK): lane holds row (lane&15); halfs j -> K=kc+hi16+j,
// hi16 = (lane>=16)*16. Two contiguous 16B loads.
__device__ __forceinline__ v16h loadA_h(const _Float16* row, int kc, int lane) {
  int hi16 = (lane >> 4) << 4;
  union { u32x4 q[2]; v16h v; } u;
  u.q[0] = *(const u32x4*)(row + kc + hi16);
  u.q[1] = *(const u32x4*)(row + kc + hi16 + 8);
  return u.v;
}
__device__ __forceinline__ v16bf loadA_bf(const unsigned short* row, int kc, int lane) {
  int hi16 = (lane >> 4) << 4;
  union { u32x4 q[2]; v16bf v; } u;
  u.q[0] = *(const u32x4*)(row + kc + hi16);
  u.q[1] = *(const u32x4*)(row + kc + hi16 + 8);
  return u.v;
}
// B fragment (32x16, KxN): lane holds column (lane&15); 16 contiguous halfs at
// off = kc + (lane>=16)*16. One 32B load.
__device__ __forceinline__ v16h loadB_h(const _Float16* colrow, int off) {
  union { u32x4 q[2]; v16h v; } u;
  const u32x4* p = (const u32x4*)(colrow + off);
  u.q[0] = p[0]; u.q[1] = p[1];
  return u.v;
}
__device__ __forceinline__ v16bf loadB_bf(const unsigned short* colrow, int off) {
  union { u32x4 q[2]; v16bf v; } u;
  const u32x4* p = (const u32x4*)(colrow + off);
  u.q[0] = p[0]; u.q[1] = p[1];
  return u.v;
}

// descending top-9; cheap reject branch, then BRANCHLESS insertion (cndmask chain)
__device__ __forceinline__ void topk_insert(float v, int c, float (&bv)[9], int (&bi)[9]) {
  if (v > bv[8]) {
    float cv = v; int ci = c;
#pragma unroll
    for (int i = 0; i < 9; ++i) {
      bool  gt = cv > bv[i];
      float nv = gt ? cv : bv[i];
      float ov = gt ? bv[i] : cv;
      int   ni = gt ? ci : bi[i];
      int   oi = gt ? bi[i] : ci;
      bv[i] = nv; bi[i] = ni; cv = ov; ci = oi;
    }
  }
}

#if USE_TDM
// Tensor Data Mover: 1-D contiguous copy of 6144 B (1536 dwords) global -> LDS.
// D# per CDNA5 ISA ch.8: group0 {count=1, lds_addr, global_addr[56:0], type=2},
// group1 {data_size=4B, tensor_dim0=1536, tensor_dim1=1, tile_dim0=1536,
//         tile_dim1=1, tensor_dim0_stride=1536}.
__device__ __forceinline__ void tdm_stage(const _Float16* gsrc, unsigned lds_off) {
  unsigned long long ga = (unsigned long long)(uintptr_t)gsrc;
  u32x4 g0;
  g0.x = 1u;                                               // count=1
  g0.y = lds_off;                                          // lds_addr
  g0.z = (unsigned)(ga & 0xFFFFFFFFu);                     // global_addr[31:0]
  g0.w = (unsigned)((ga >> 32) & 0x01FFFFFFu) | (2u << 30);// addr[56:32] | type=2
  i32x8 g1;
  g1[0] = (int)(2u << 16);        // workgroup_mask=0, data_size=2 (4-byte)
  g1[1] = (int)(1536u << 16);     // tensor_dim0[15:0] (bits 63:48)
  g1[2] = (int)(1u << 16);        // tensor_dim0 hi=0 | tensor_dim1=1
  g1[3] = (int)(1536u << 16);     // tensor_dim1 hi=0 | tile_dim0=1536
  g1[4] = 1;                      // tile_dim1=1, tile_dim2=0
  g1[5] = 1536;                   // tensor_dim0_stride[31:0]
  g1[6] = 0;
  g1[7] = 0;
  i32x4 z4 = {0, 0, 0, 0};
  i32x8 z8 = {0, 0, 0, 0, 0, 0, 0, 0};
  // 6-arg variant (therock-10.0 headers / clang-23 toolchain)
  __builtin_amdgcn_tensor_load_to_lds(g0, g1, z4, z4, z8, 0);
}
#endif

// ---------- kernel 1: transpose + norms -> xn (f16, normalized) and xt (bf16) ----------
__global__ __launch_bounds__(256) void k_prep_x(const float* __restrict__ x,
                                                _Float16* __restrict__ xn,
                                                unsigned short* __restrict__ xt) {
  int g = blockIdx.x * 256 + threadIdx.x;          // 0..Rc-1
  int b = g / Nc, n = g % Nc;
  const float* src = x + (size_t)b * Cc * Nc + n;  // stride Nc over c
  unsigned short* drow = xt + (size_t)g * Cc;
  float ss = 0.f;
  for (int c = 0; c < Cc; ++c) {
    float v = src[(size_t)c * Nc];
    ss += v * v;
    drow[c] = f2bf(v);
  }
  float inv = 1.0f / fmaxf(sqrtf(ss), 1e-12f);
  _Float16* nrow = xn + (size_t)g * Cc;
  for (int c = 0; c < Cc; ++c) {
    float v = src[(size_t)c * Nc];
    nrow[c] = (_Float16)(v * inv);
  }
}

// ---------- kernel 2: conv_w -> bf16 ----------
__global__ __launch_bounds__(256) void k_prep_w(const float* __restrict__ w,
                                                unsigned short* __restrict__ wbf) {
  int i = blockIdx.x * 256 + threadIdx.x;
  if (i < OUTc * Ec) wbf[i] = f2bf(w[i]);
}

// ---------- kernel 3: fused Gram-GEMM (WMMA f16) + streaming top-9 -> idx ----------
// grid (49, B), block 128 (4 waves). Wave w owns row tile rt = blockIdx.x*4+w.
// B tiles staged by the Tensor Data Mover, double-buffered, overlapped with WMMA.
__global__ __launch_bounds__(128) void k_knn(const _Float16* __restrict__ xn,
                                             int* __restrict__ idxOut) {
  __shared__ _Float16 Bs[2][16][192];     // double-buffered column-block rows (12 KB)
  __shared__ float Gt[4][16][17];         // per-wave 16x16 tile transpose scratch

  int tid  = threadIdx.x;
  int lane = tid & 31;
  int wv   = tid >> 5;
  int b    = blockIdx.y;
  int rt   = blockIdx.x * 4 + wv;         // 0..195
  int m0   = rt * 16;

  const _Float16* batchBase = xn + (size_t)b * Nc * Cc;
  const _Float16* Arow = batchBase + (size_t)(m0 + (lane & 15)) * Cc;
  v16h a[6];
#pragma unroll
  for (int kk = 0; kk < 6; ++kk) a[kk] = loadA_h(Arow, kk * 32, lane);

  float bv[9]; int bi[9];
#pragma unroll
  for (int j = 0; j < 9; ++j) { bv[j] = -1e30f; bi[j] = 0; }

  int rowOwn = lane & 15;                 // row within tile this lane tracks
  int colOwn = (lane >> 4) << 3;          // 8-column half this lane scans
  int hi16   = (lane >> 4) << 4;

#if USE_TDM
  if (wv == 0)
    tdm_stage(batchBase, (unsigned)(uintptr_t)&Bs[0][0][0]);   // prologue: tile 0
#endif

  for (int ct = 0; ct < 196; ++ct) {
#if USE_TDM
    if (wv == 0) {
      if (ct + 1 < 196) {
        tdm_stage(batchBase + (size_t)(ct + 1) * 16 * Cc,
                  (unsigned)(uintptr_t)&Bs[(ct + 1) & 1][0][0]);
        __builtin_amdgcn_s_wait_tensorcnt(1);   // current tile done, next in flight
      } else {
        __builtin_amdgcn_s_wait_tensorcnt(0);
      }
    }
#else
    __syncthreads();   // protect Bs from previous iteration's readers
    {
      const u32x4* src = (const u32x4*)(batchBase + (size_t)ct * 16 * Cc);
      u32x4* dst = (u32x4*)&Bs[ct & 1][0][0];
      for (int i = tid; i < 384; i += 128) dst[i] = src[i];
    }
#endif
    __syncthreads();   // staged tile visible to all waves

    // preload all 6 B fragments, then feed the WMMA chain
    const _Float16* brow = &Bs[ct & 1][lane & 15][0];
    v16h bfr[6];
#pragma unroll
    for (int kk = 0; kk < 6; ++kk) bfr[kk] = loadB_h(brow, kk * 32 + hi16);
    v8f acc = {};
#pragma unroll
    for (int kk = 0; kk < 6; ++kk)
      acc = __builtin_amdgcn_wmma_f32_16x16x32_f16(false, a[kk], false, bfr[kk],
                                                   (short)0, acc, false, false);

    // per-wave transpose via private LDS scratch; DS ops are in-order within a
    // wave, so no barrier is needed between these stores and loads.
#pragma unroll
    for (int v = 0; v < 8; ++v)
      Gt[wv][v + ((lane >> 4) << 3)][lane & 15] = acc[v];

    int cbase = ct * 16 + colOwn;
#pragma unroll
    for (int cc = 0; cc < 8; ++cc)
      topk_insert(Gt[wv][rowOwn][colOwn + cc], cbase + cc, bv, bi);

    __syncthreads();   // all waves done reading Bs[ct&1] before TDM reuses it
  }

  // merge lane pair (lane, lane^16) which share a row
#pragma unroll
  for (int j = 0; j < 9; ++j) {
    float pv = __shfl_xor(bv[j], 16);
    int   pi = __shfl_xor(bi[j], 16);
    topk_insert(pv, pi, bv, bi);
  }
  if (lane < 16) {
    int* out = idxOut + ((size_t)b * Nc + m0 + lane) * 9;
#pragma unroll
    for (int j = 0; j < 9; ++j) out[j] = bi[j];
  }
}

// ---------- kernel 4: P = X*W1^T, Q = X*W2^T (WMMA bf16) ----------
// grid 1568 row tiles, block 128 (4 waves); wave handles o-tiles {wv, wv+4, wv+8}
__global__ __launch_bounds__(128) void k_pqgemm(const unsigned short* __restrict__ xt,
                                                const unsigned short* __restrict__ wbf,
                                                float* __restrict__ P,
                                                float* __restrict__ Q) {
  int lane = threadIdx.x & 31;
  int wv   = threadIdx.x >> 5;
  int g0   = blockIdx.x * 16;

  const unsigned short* Arow = xt + (size_t)(g0 + (lane & 15)) * Cc;
  v16bf a[6];
#pragma unroll
  for (int kk = 0; kk < 6; ++kk) a[kk] = loadA_bf(Arow, kk * 32, lane);

  for (int ot = wv; ot < 12; ot += 4) {
    int o0 = ot * 16;
    const unsigned short* Wrow = wbf + (size_t)(o0 + (lane & 15)) * Ec;
    v8f ap = {}, aq = {};
#pragma unroll
    for (int kk = 0; kk < 6; ++kk) {
      int off = kk * 32 + ((lane >> 4) << 4);
      v16bf bp = loadB_bf(Wrow, off);            // W1 half (cols 0..191)
      v16bf bq = loadB_bf(Wrow + Cc, off);       // W2 half (cols 192..383)
      ap = __builtin_amdgcn_wmma_f32_16x16x32_bf16(false, a[kk], false, bp,
                                                   (short)0, ap, false, false);
      aq = __builtin_amdgcn_wmma_f32_16x16x32_bf16(false, a[kk], false, bq,
                                                   (short)0, aq, false, false);
    }
#pragma unroll
    for (int v = 0; v < 8; ++v) {
      int m = v + ((lane >> 4) << 3);
      size_t o = (size_t)(g0 + m) * OUTc + o0 + (lane & 15);
      P[o] = ap[v];
      Q[o] = aq[v];
    }
  }
}

// ---------- kernel 5: BN partial sums per channel (deterministic) ----------
__global__ __launch_bounds__(192) void k_pass1(const float* __restrict__ P,
                                               const float* __restrict__ Q,
                                               const int* __restrict__ idx,
                                               const float* __restrict__ bias,
                                               float* __restrict__ partial) {
  int o = threadIdx.x;
  __shared__ int sj[9];
  float s1 = 0.f, s2 = 0.f;
  float bo = bias[o];
  for (int r = blockIdx.x; r < Rc; r += gridDim.x) {
    if (o < 9) sj[o] = idx[(size_t)r * 9 + o];
    __syncthreads();
    int b = r / Nc;
    float base = P[(size_t)r * OUTc + o] + bo - Q[(size_t)r * OUTc + o];
    size_t bofs = (size_t)b * Nc;
#pragma unroll
    for (int k = 0; k < 9; ++k) {
      float y = base + Q[(bofs + sj[k]) * OUTc + o];
      s1 += y; s2 += y * y;
    }
    __syncthreads();
  }
  partial[(size_t)blockIdx.x * 384 + o]       = s1;
  partial[(size_t)blockIdx.x * 384 + 192 + o] = s2;
}

__global__ __launch_bounds__(384) void k_reduce(const float* __restrict__ partial,
                                                float* __restrict__ stats) {
  int t = threadIdx.x;
  float s = 0.f;
  for (int blk = 0; blk < 256; ++blk) s += partial[(size_t)blk * 384 + t];
  stats[t] = s;
}

// ---------- kernel 6: BN + exact GELU + max over K + transpose ----------
// grid Rc/32 = 784, block 192. Block handles 32 rows x all 192 channels.
__global__ __launch_bounds__(192) void k_pass2(const float* __restrict__ P,
                                               const float* __restrict__ Q,
                                               const int* __restrict__ idx,
                                               const float* __restrict__ bias,
                                               const float* __restrict__ gamma,
                                               const float* __restrict__ beta,
                                               const float* __restrict__ stats,
                                               float* __restrict__ out) {
  int o  = threadIdx.x;
  int g0 = blockIdx.x * 32;
  int b  = g0 / Nc;                       // 3136 % 32 == 0: never crosses batch
  int n0 = g0 % Nc;

  __shared__ int   sj[32][9];
  __shared__ float ymax[192][33];         // 33 pad -> conflict-free

  for (int i = o; i < 32 * 9; i += 192)
    sj[i / 9][i % 9] = idx[(size_t)g0 * 9 + i];

  const float CNTf = (float)(Bc) * (float)(Nc) * (float)(Kc);  // 225792
  float mean = stats[o] / CNTf;
  float var  = stats[192 + o] / CNTf - mean * mean;
  float scale = gamma[o] * rsqrtf(var + 1e-5f);
  float shift = beta[o] - mean * scale;
  float bo = bias[o];
  size_t bofs = (size_t)b * Nc;
  __syncthreads();

  for (int n = 0; n < 32; ++n) {
    size_t r = (size_t)(g0 + n);
    float base = P[r * OUTc + o] + bo - Q[r * OUTc + o];
    float m = -3.4e38f;
#pragma unroll
    for (int k = 0; k < 9; ++k) {
      float y  = base + Q[(bofs + sj[n][k]) * OUTc + o];
      float yb = y * scale + shift;
      float g  = 0.5f * yb * (1.0f + erff(yb * 0.70710678118654752f));
      m = fmaxf(m, g);
    }
    ymax[o][n] = m;
  }
  __syncthreads();

  // coalesced transposed store: out[b][oo][n0+nn]
  for (int i = o; i < 192 * 32; i += 192) {
    int oo = i >> 5, nn = i & 31;
    out[((size_t)b * OUTc + oo) * Nc + n0 + nn] = ymax[oo][nn];
  }
}

// ---------- host launch ----------
extern "C" void kernel_launch(void* const* d_in, const int* in_sizes, int n_in,
                              void* d_out, int out_size, void* d_ws, size_t ws_size,
                              hipStream_t stream) {
  (void)in_sizes; (void)n_in; (void)out_size; (void)ws_size;
  const float* x      = (const float*)d_in[0];
  const float* conv_w = (const float*)d_in[1];
  const float* conv_b = (const float*)d_in[2];
  const float* gamma  = (const float*)d_in[3];
  const float* beta   = (const float*)d_in[4];
  float* out = (float*)d_out;

  char* p = (char*)d_ws;
  _Float16*       xn      = (_Float16*)p;       p += (size_t)Rc * Cc * 2;
  unsigned short* xt      = (unsigned short*)p; p += (size_t)Rc * Cc * 2;
  unsigned short* wbf     = (unsigned short*)p; p += (size_t)OUTc * Ec * 2;
  int*            idx     = (int*)p;            p += (size_t)Rc * Kc * 4;
  float*          Pm      = (float*)p;          p += (size_t)Rc * OUTc * 4;
  float*          Qm      = (float*)p;          p += (size_t)Rc * OUTc * 4;
  float*          partial = (float*)p;          p += (size_t)256 * 384 * 4;
  float*          stats   = (float*)p;          p += (size_t)384 * 4;

  k_prep_x<<<Rc / 256, 256, 0, stream>>>(x, xn, xt);
  k_prep_w<<<(OUTc * Ec + 255) / 256, 256, 0, stream>>>(conv_w, wbf);
  k_knn<<<dim3(49, Bc), 128, 0, stream>>>(xn, idx);
  k_pqgemm<<<Rc / 16, 128, 0, stream>>>(xt, wbf, Pm, Qm);
  k_pass1<<<256, 192, 0, stream>>>(Pm, Qm, idx, conv_b, partial);
  k_reduce<<<1, 384, 0, stream>>>(partial, stats);
  k_pass2<<<Rc / 32, 192, 0, stream>>>(Pm, Qm, idx, conv_b, gamma, beta, stats, out);
}
